// TripletMinner_67920612819124
// MI455X (gfx1250) — 8-bit matrix operand order repaired and checked by reading, compiled, not run on hardware
//
#include <hip/hip_runtime.h>

typedef __attribute__((ext_vector_type(8))) int   v8i;
typedef __attribute__((ext_vector_type(4))) float v4f;

#define BATCH 512
#define NCLS  100
#define KPAD  128

__global__ void cvt_labels_u8(const float* __restrict__ labels,
                              unsigned char* __restrict__ lab8) {
    const int i = blockIdx.x;
    const int d = threadIdx.x;
    unsigned int packed = 0;
#pragma unroll
    for (int b = 0; b < 4; ++b) {
        const int k = d * 4 + b;
        unsigned int v = 0;
        if (k < NCLS) v = (unsigned int)labels[i * NCLS + k];
        packed |= (v & 0xFFu) << (8 * b);
    }
    ((unsigned int*)lab8)[i * (KPAD / 4) + d] = packed;
}

__global__ void matches_wmma(const unsigned char* __restrict__ lab8,
                             unsigned char* __restrict__ M) {
    const unsigned int* labw = (const unsigned int*)lab8;
    const int L  = threadIdx.x;
    const int h  = L >> 4;
    const int ln = L & 15;

    const int arow_g = blockIdx.y * 16 + ln;
    const int bcol_g = blockIdx.x * 16 + ln;

    const unsigned int* arow = labw + arow_g * (KPAD / 4);
    const unsigned int* brow = labw + bcol_g * (KPAD / 4);

    const int a_idx[8] = { h*2+0, h*2+1, 4+h*2, 5+h*2,
                           8+h*2, 9+h*2, 12+h*2, 13+h*2 };
    const int b_idx[8] = { h*2+0, h*2+1, 4+h*2, 5+h*2,
                           8+h*2, 9+h*2, 12+h*2, 13+h*2 };

    v8i a0, a1, b0, b1;
#pragma unroll
    for (int v = 0; v < 8; ++v) {
        a0[v] = (int)arow[a_idx[v]];
        a1[v] = (int)arow[a_idx[v] + 16];
        b0[v] = (int)brow[b_idx[v]];
        b1[v] = (int)brow[b_idx[v] + 16];
    }

    v8i acc = {};
    acc = __builtin_amdgcn_wmma_i32_16x16x64_iu8(false, a0, false, b0, acc,
                                                 false, false);
    acc = __builtin_amdgcn_wmma_i32_16x16x64_iu8(false, a1, false, b1, acc,
                                                 false, false);

#pragma unroll
    for (int r = 0; r < 8; ++r) {
        const int gm = blockIdx.y * 16 + r + 8 * h;
        M[gm * BATCH + bcol_g] = (unsigned char)acc[r];
    }
}

#define JCHUNK 64
__global__ void cube_kernel(const unsigned char* __restrict__ M,
                            float* __restrict__ out) {
    __shared__ unsigned char rowm[BATCH];
    const int i     = blockIdx.y;
    const int jbase = blockIdx.x * JCHUNK;
    const int tid   = threadIdx.x;

    ((unsigned int*)rowm)[tid] = ((const unsigned int*)(M + i * BATCH))[tid];
    __syncthreads();

    const unsigned int mk = ((unsigned int*)rowm)[tid];
    const unsigned int dk = mk ^ 0x01010101u;
    const unsigned int d0 =  dk        & 0xFFu;
    const unsigned int d1 = (dk >> 8)  & 0xFFu;
    const unsigned int d2 = (dk >> 16) & 0xFFu;
    const unsigned int d3 = (dk >> 24) & 0xFFu;

    float* outp = out + ((size_t)i * BATCH + jbase) * BATCH + tid * 4;
#pragma unroll 4
    for (int j = 0; j < JCHUNK; ++j) {
        const unsigned int mj = rowm[jbase + j];
        v4f v;
        v.x = (float)((mj * d0) & 0xFFu);
        v.y = (float)((mj * d1) & 0xFFu);
        v.z = (float)((mj * d2) & 0xFFu);
        v.w = (float)((mj * d3) & 0xFFu);
        __builtin_nontemporal_store(v, (v4f*)outp);
        outp += BATCH;
    }
}

extern "C" void kernel_launch(void* const* d_in, const int* in_sizes, int n_in,
                              void* d_out, int out_size, void* d_ws, size_t ws_size,
                              hipStream_t stream) {
    const float* labels = (const float*)d_in[0];
    float*       out    = (float*)d_out;

    unsigned char* lab8 = (unsigned char*)d_ws;
    unsigned char* M    = lab8 + (size_t)BATCH * KPAD;

    cvt_labels_u8<<<dim3(BATCH), dim3(32), 0, stream>>>(labels, lab8);
    matches_wmma<<<dim3(BATCH / 16, BATCH / 16), dim3(32), 0, stream>>>(lab8, M);
    cube_kernel<<<dim3(BATCH / JCHUNK, BATCH), dim3(128), 0, stream>>>(M, out);
}
